// GroupedQueryAttention_6992206758373
// MI455X (gfx1250) — matrix operand order repaired and checked by reading, not compiled, not run on hardware
//
#include <hip/hip_runtime.h>
#include <hip/hip_bf16.h>

typedef __bf16 bf16;
typedef __attribute__((ext_vector_type(16))) __bf16 v16bf;
typedef __attribute__((ext_vector_type(8)))  __bf16 v8bf;
typedef __attribute__((ext_vector_type(8)))  float  v8f;
typedef __attribute__((ext_vector_type(4))) unsigned int u32x4;
typedef __attribute__((ext_vector_type(8))) int         i32x8;
typedef __attribute__((ext_vector_type(4))) int         i32x4;

#define B_    2
#define S_    2048
#define DIM_  2048
#define NKV_  8
#define NG_   4
#define HD_   64
#define MTOT  (B_ * S_)     // 4096
#define KVD   (NKV_ * HD_)  // 512

// ---------------------------------------------------------------------------
// WMMA helper (CDNA5 16x16x32 bf16, fp32 accumulate)
// ---------------------------------------------------------------------------
__device__ __forceinline__ v8f wmma_bf16(v16bf a, v16bf b, v8f c) {
  return __builtin_amdgcn_wmma_f32_16x16x32_bf16(false, a, false, b,
                                                 (short)0, c, false, false);
}

// A fragment (16x32 MxK): lane half -> kb = 0/8; two contiguous 8-elem runs.
__device__ __forceinline__ v16bf frag_a(const bf16* base, int ld, int lane) {
  const int m  = lane & 15;
  const int kb = (lane & 16) ? 8 : 0;
  const bf16* p = base + m * ld + kb;
  union { v16bf v; v8bf h[2]; } u;
  u.h[0] = *(const v8bf*)(p);
  u.h[1] = *(const v8bf*)(p + 16);
  return u.v;
}

// B fragment (32x16 KxN): lane n = lane&15, lane half -> kb = 0/8; two contiguous 8-elem runs.
__device__ __forceinline__ v16bf frag_b(const bf16* base, int ld, int lane) {
  const int n  = lane & 15;
  const int kb = (lane & 16) ? 8 : 0;
  const bf16* p = base + n * ld + kb;
  union { v16bf v; v8bf h[2]; } u;
  u.h[0] = *(const v8bf*)(p);
  u.h[1] = *(const v8bf*)(p + 16);
  return u.v;
}

// ---------------------------------------------------------------------------
// TDM: 2D tensor tile -> LDS with row padding (CDNA5 ISA).
// D# group0: count=1 | lds_addr | global_addr | type=2
// D# group1: wgmask=0, data_size=1(2B), pad_enable, pad codes, dims/strides.
// This toolchain's builtin takes 6 args (extra int32x8 group before cpol).
// ---------------------------------------------------------------------------
__device__ __forceinline__ unsigned int lds_off(const void* p) {
  // generic LDS pointer = {shared aperture, offset[31:0]}
  return (unsigned int)(unsigned long long)p;
}

__device__ __forceinline__ void tdm_load_2d(unsigned int lds_byte_off,
                                            const void* gptr,
                                            unsigned int tensor_d0,
                                            unsigned int tensor_d1,
                                            unsigned int tile_d0,
                                            unsigned int tile_d1,
                                            unsigned long long stride0,
                                            unsigned int pad_interval_code,
                                            unsigned int pad_amount_code) {
  const unsigned long long ga = (unsigned long long)gptr;
  u32x4 g0;
  g0.x = 1u;                                   // count=1 (valid), user mode
  g0.y = lds_byte_off;                         // bits [63:32]
  g0.z = (unsigned int)ga;                     // global addr lo
  g0.w = (unsigned int)(ga >> 32) | (2u << 30);// global addr hi | type=2
  const unsigned int d0 = (1u << 16)           // data_size = 2 bytes
                        | (1u << 20)           // pad_enable
                        | (pad_interval_code << 22)
                        | (pad_amount_code << 25);
  const unsigned int d1 = (tensor_d0 & 0xFFFFu) << 16;       // dim0 lo16 @bit48
  const unsigned int d2 = (tensor_d0 >> 16) | ((tensor_d1 & 0xFFFFu) << 16);
  const unsigned int d3 = (tensor_d1 >> 16) | (tile_d0 << 16);
  const unsigned int d4 = tile_d1;                           // tile_dim2 = 0
  const unsigned int d5 = (unsigned int)stride0;             // stride0 @bit160
  const unsigned int d6 = (unsigned int)((stride0 >> 32) & 0xFFFFull);
  const unsigned int d7 = 0;                                 // stride1 = 0
  i32x8 g1 = {(int)d0, (int)d1, (int)d2, (int)d3,
              (int)d4, (int)d5, (int)d6, (int)d7};
  i32x4 z4 = {0, 0, 0, 0};
  i32x8 z8 = {0, 0, 0, 0, 0, 0, 0, 0};
  __builtin_amdgcn_tensor_load_to_lds(g0, g1, z4, z4, z8, 0);
}

// ---------------------------------------------------------------------------
// fp32 -> bf16 conversion (plain and transposing)
// ---------------------------------------------------------------------------
__global__ void cvt_f32_bf16(const float* __restrict__ src,
                             bf16* __restrict__ dst, int n) {
  int i = blockIdx.x * blockDim.x + threadIdx.x;
  const int stride = gridDim.x * blockDim.x;
  for (; i < n; i += stride) dst[i] = (bf16)src[i];
}

// src[K][N] fp32 -> dst[N][K] bf16 (weights become K-contiguous per column)
__global__ void cvt_f32_bf16_T(const float* __restrict__ src,
                               bf16* __restrict__ dst, int K, int N) {
  int i = blockIdx.x * blockDim.x + threadIdx.x;
  const int total = K * N;
  const int stride = gridDim.x * blockDim.x;
  for (; i < total; i += stride) {
    const int k = i / N, n = i - k * N;
    dst[(size_t)n * K + k] = (bf16)src[i];
  }
}

// ---------------------------------------------------------------------------
// GEMM: C[M,N] = A[M,K] (row major) * WT[N,K] (row major, i.e. W transposed)
// Block tile 128x128, BK=32, double-buffered LDS filled by TDM; 8 waves in
// 2(m) x 4(n); each wave computes 64x32 = 4x2 WMMA tiles (8 wmma / K-step).
// ---------------------------------------------------------------------------
template <bool OUT_F32>
__global__ __launch_bounds__(256)
void gemm_bf16(const bf16* __restrict__ A, const bf16* __restrict__ WT,
               void* __restrict__ Cout, int M, int N, int K) {
  constexpr int BM = 128, BN = 128, BK = 32;
  constexpr int LD = BK + 8;  // 40 elems = 80B rows (TDM pad: 16DW + 4DW)

  __shared__ __align__(16) bf16 As[2][BM * LD];
  __shared__ __align__(16) bf16 Bs[2][BN * LD];

  const int tid  = threadIdx.x;
  const int lane = tid & 31;
  const int w    = tid >> 5;   // 0..7
  const int wm   = w & 1;      // 2 waves in m
  const int wn   = w >> 1;     // 4 waves in n
  const int m0   = blockIdx.y * BM;
  const int n0   = blockIdx.x * BN;

  v8f acc[4][2] = {};

  // pad codes: 16 DWORDs interval (code 3), 4 DWORDs pad (code 3) -> LD=40
  auto issue = [&](int buf, int k0) {
    if (w == 0) {
      tdm_load_2d(lds_off(&As[buf][0]), A + (size_t)m0 * K + k0,
                  (unsigned)K, (unsigned)M, BK, BM, (unsigned long long)K, 3, 3);
      tdm_load_2d(lds_off(&Bs[buf][0]), WT + (size_t)n0 * K + k0,
                  (unsigned)K, (unsigned)N, BK, BN, (unsigned long long)K, 3, 3);
    }
  };

  issue(0, 0);
  if (w == 0) __builtin_amdgcn_s_wait_tensorcnt(0);
  __syncthreads();

  int buf = 0;
  for (int k0 = 0; k0 < K; k0 += BK) {
    const bool more = (k0 + BK) < K;
    if (more) issue(buf ^ 1, k0 + BK);  // overlap TDM with WMMA

    v16bf a[4], b[2];
#pragma unroll
    for (int mt = 0; mt < 4; ++mt)
      a[mt] = frag_a(&As[buf][(wm * 64 + mt * 16) * LD], LD, lane);
#pragma unroll
    for (int nt = 0; nt < 2; ++nt)
      b[nt] = frag_b(&Bs[buf][(wn * 32 + nt * 16) * LD], LD, lane);
#pragma unroll
    for (int mt = 0; mt < 4; ++mt)
#pragma unroll
      for (int nt = 0; nt < 2; ++nt)
        acc[mt][nt] = wmma_bf16(a[mt], b[nt], acc[mt][nt]);

    if (w == 0 && more) __builtin_amdgcn_s_wait_tensorcnt(0);
    __syncthreads();
    buf ^= 1;
  }

  const int hi = (lane >> 4) & 1;
  const int nl = lane & 15;
#pragma unroll
  for (int mt = 0; mt < 4; ++mt)
#pragma unroll
    for (int nt = 0; nt < 2; ++nt)
#pragma unroll
      for (int r = 0; r < 8; ++r) {
        const int row = m0 + wm * 64 + mt * 16 + r + hi * 8;
        const int col = n0 + wn * 32 + nt * 16 + nl;
        const float vv = acc[mt][nt][r];
        if (OUT_F32)
          ((float*)Cout)[(size_t)row * N + col] = vv;
        else
          ((bf16*)Cout)[(size_t)row * N + col] = (bf16)vv;
      }
}

// ---------------------------------------------------------------------------
// Flash attention, 64-key blocks. K block staged by TDM (row-major, padded),
// V staged manually transposed. Row sums via WMMA-with-ones (no shuffles);
// only the row max uses 4 ds_bpermute steps.
// ---------------------------------------------------------------------------
__global__ __launch_bounds__(128)
void flash_attn(const bf16* __restrict__ Q, const bf16* __restrict__ K,
                const bf16* __restrict__ V, bf16* __restrict__ O) {
  constexpr int KB  = 64;
  constexpr int KLD = HD_ + 8;  // 72 elems = 144B rows (TDM pad: 32DW + 4DW)
  constexpr int VLD = KB + 8;   // 72
  constexpr int PLD = KB + 8;   // 72

  __shared__ __align__(16) bf16 Ks[KB * KLD];     // [key][d]
  __shared__ __align__(16) bf16 VsT[HD_ * VLD];   // [d][key]
  __shared__ __align__(16) bf16 Ps[4][16 * PLD];  // per-wave P staging

  const int tid  = threadIdx.x;
  const int lane = tid & 31;
  const int w    = tid >> 5;  // 0..3
  const int b    = blockIdx.z;
  const int hq   = blockIdx.y;   // 0..31, q col = hq*64
  const int hkv  = hq >> 2;
  const int q0   = blockIdx.x * 64 + w * 16;

  const bf16* qptr = Q + (size_t)b * S_ * DIM_ + (size_t)hq * HD_;
  const bf16* kptr = K + (size_t)b * S_ * KVD + (size_t)hkv * HD_;
  const bf16* vptr = V + (size_t)b * S_ * KVD + (size_t)hkv * HD_;
  bf16*       optr = O + (size_t)b * S_ * DIM_ + (size_t)hq * HD_;

  const int ml = lane & 15;
  const int hi = (lane >> 4) & 1;

  // Q A-fragments (d = 0..31, 32..63) kept in registers
  v16bf aq[2];
#pragma unroll
  for (int ks = 0; ks < 2; ++ks) {
    const bf16* p = qptr + (size_t)(q0 + ml) * DIM_ + ks * 32 + (hi ? 8 : 0);
    union { v16bf v; v8bf h[2]; } u;
    u.h[0] = *(const v8bf*)(p);
    u.h[1] = *(const v8bf*)(p + 16);
    aq[ks] = u.v;
  }

  v16bf ones;
#pragma unroll
  for (int i = 0; i < 16; ++i) ones[i] = (bf16)1.0f;

  float mrow[8], lrow[8], alpha[8];
  v8f o[4] = {};
#pragma unroll
  for (int r = 0; r < 8; ++r) { mrow[r] = -1e30f; lrow[r] = 0.0f; }
  const float sm = 0.125f;  // HEAD_DIM^-0.5

  for (int kb0 = 0; kb0 < S_; kb0 += KB) {
    // ---- K block [64 keys][64 d] via TDM, padded rows of 72 ----
    if (w == 0)
      tdm_load_2d(lds_off(Ks), kptr + (size_t)kb0 * KVD,
                  (unsigned)KVD, (unsigned)S_, HD_, KB,
                  (unsigned long long)KVD, 4, 3);
    // ---- V block transposed to [d][key] (manual; TDM can't transpose) ----
#pragma unroll
    for (int it = 0; it < 4; ++it) {
      const int c   = tid + it * 128;   // 0..511
      const int row = c >> 3;           // key 0..63
      const int col = (c & 7) * 8;      // d
      v8bf vv = *(const v8bf*)(vptr + (size_t)(kb0 + row) * KVD + col);
#pragma unroll
      for (int j = 0; j < 8; ++j) VsT[(col + j) * VLD + row] = vv[j];
    }
    if (kb0 + KB < S_)  // prefetch next V block (global_prefetch_b8)
      __builtin_prefetch(vptr + (size_t)(kb0 + KB + (tid >> 1)) * KVD, 0, 0);
    if (w == 0) __builtin_amdgcn_s_wait_tensorcnt(0);
    __syncthreads();

    // ---- scores: 4 key n-tiles x 2 K-steps over head dim ----
    v8f s[4] = {};
#pragma unroll
    for (int j = 0; j < 4; ++j) {
      s[j] = wmma_bf16(aq[0], frag_b(&Ks[(j * 16) * KLD + 0], KLD, lane), s[j]);
      s[j] = wmma_bf16(aq[1], frag_b(&Ks[(j * 16) * KLD + 32], KLD, lane), s[j]);
    }

    // ---- online softmax: max via 4 shuffles; exp; stage P to LDS ----
    bf16* pw = &Ps[w][0];
#pragma unroll
    for (int r = 0; r < 8; ++r) {
      float v0 = s[0][r] * sm, v1 = s[1][r] * sm;
      float v2 = s[2][r] * sm, v3 = s[3][r] * sm;
      float bmx = fmaxf(fmaxf(v0, v1), fmaxf(v2, v3));
      bmx = fmaxf(bmx, __shfl_xor(bmx, 1, 32));
      bmx = fmaxf(bmx, __shfl_xor(bmx, 2, 32));
      bmx = fmaxf(bmx, __shfl_xor(bmx, 4, 32));
      bmx = fmaxf(bmx, __shfl_xor(bmx, 8, 32));
      const float mn = fmaxf(mrow[r], bmx);
      alpha[r] = __expf(mrow[r] - mn);
      mrow[r] = mn;
      const int row = r + hi * 8;
      pw[row * PLD + 0 * 16 + ml] = (bf16)__expf(v0 - mn);
      pw[row * PLD + 1 * 16 + ml] = (bf16)__expf(v1 - mn);
      pw[row * PLD + 2 * 16 + ml] = (bf16)__expf(v2 - mn);
      pw[row * PLD + 3 * 16 + ml] = (bf16)__expf(v3 - mn);
#pragma unroll
      for (int dt = 0; dt < 4; ++dt) o[dt][r] *= alpha[r];
    }

    // ---- P fragments (same-wave DS ops are in-order) ----
    const v16bf pa0 = frag_a(pw, PLD, lane);       // keys 0..31
    const v16bf pa1 = frag_a(pw + 32, PLD, lane);  // keys 32..63

    // ---- row sums via P @ ones (broadcast to every lane) ----
    v8f sums = {};
    sums = wmma_bf16(pa0, ones, sums);
    sums = wmma_bf16(pa1, ones, sums);
#pragma unroll
    for (int r = 0; r < 8; ++r) lrow[r] = lrow[r] * alpha[r] + sums[r];

    // ---- O += P(16x64) @ V(64x64) ----
#pragma unroll
    for (int dt = 0; dt < 4; ++dt) {
      o[dt] = wmma_bf16(pa0, frag_b(&VsT[(dt * 16) * VLD + 0], VLD, lane), o[dt]);
      o[dt] = wmma_bf16(pa1, frag_b(&VsT[(dt * 16) * VLD + 32], VLD, lane), o[dt]);
    }

    __syncthreads();
  }

  // ---- normalize + store ----
#pragma unroll
  for (int r = 0; r < 8; ++r) {
    const float inv = 1.0f / lrow[r];
    const int row = q0 + r + hi * 8;
#pragma unroll
    for (int dt = 0; dt < 4; ++dt)
      optr[(size_t)row * DIM_ + dt * 16 + ml] = (bf16)(o[dt][r] * inv);
  }
}

// ---------------------------------------------------------------------------
// Host-side orchestration
// ---------------------------------------------------------------------------
extern "C" void kernel_launch(void* const* d_in, const int* in_sizes, int n_in,
                              void* d_out, int out_size, void* d_ws,
                              size_t ws_size, hipStream_t stream) {
  (void)in_sizes; (void)n_in; (void)out_size; (void)ws_size;
  const float* x  = (const float*)d_in[0];
  const float* Wq = (const float*)d_in[1];
  const float* Wk = (const float*)d_in[2];
  const float* Wv = (const float*)d_in[3];
  const float* Wo = (const float*)d_in[4];

  char* p = (char*)d_ws;
  auto take = [&](size_t bytes) {
    char* r = p;
    p += (bytes + 255) & ~(size_t)255;
    return r;
  };
  bf16* xb   = (bf16*)take((size_t)MTOT * DIM_ * 2);
  bf16* wqbT = (bf16*)take((size_t)DIM_ * DIM_ * 2);  // [N][K]
  bf16* wkbT = (bf16*)take((size_t)DIM_ * KVD * 2);
  bf16* wvbT = (bf16*)take((size_t)DIM_ * KVD * 2);
  bf16* wobT = (bf16*)take((size_t)DIM_ * DIM_ * 2);
  bf16* qb   = (bf16*)take((size_t)MTOT * DIM_ * 2);
  bf16* kb   = (bf16*)take((size_t)MTOT * KVD * 2);
  bf16* vb   = (bf16*)take((size_t)MTOT * KVD * 2);
  bf16* ob   = (bf16*)take((size_t)MTOT * DIM_ * 2);

  cvt_f32_bf16<<<2048, 256, 0, stream>>>(x, xb, MTOT * DIM_);
  cvt_f32_bf16_T<<<2048, 256, 0, stream>>>(Wq, wqbT, DIM_, DIM_);
  cvt_f32_bf16_T<<<512, 256, 0, stream>>>(Wk, wkbT, DIM_, KVD);
  cvt_f32_bf16_T<<<512, 256, 0, stream>>>(Wv, wvbT, DIM_, KVD);
  cvt_f32_bf16_T<<<2048, 256, 0, stream>>>(Wo, wobT, DIM_, DIM_);

  // Projections (bf16 out)
  gemm_bf16<false><<<dim3(DIM_ / 128, MTOT / 128), 256, 0, stream>>>(
      xb, wqbT, qb, MTOT, DIM_, DIM_);
  gemm_bf16<false><<<dim3(KVD / 128, MTOT / 128), 256, 0, stream>>>(
      xb, wkbT, kb, MTOT, KVD, DIM_);
  gemm_bf16<false><<<dim3(KVD / 128, MTOT / 128), 256, 0, stream>>>(
      xb, wvbT, vb, MTOT, KVD, DIM_);

  // Attention
  flash_attn<<<dim3(S_ / 64, 32, B_), 128, 0, stream>>>(qb, kb, vb, ob);

  // Output projection (fp32 out -> d_out)
  gemm_bf16<true><<<dim3(DIM_ / 128, MTOT / 128), 256, 0, stream>>>(
      ob, wobT, d_out, MTOT, DIM_, DIM_);
}
